// MoeLayer_5093831213777
// MI455X (gfx1250) — matrix operand order repaired and checked by reading, not compiled, not run on hardware
//
#include <hip/hip_runtime.h>
#include <hip/hip_bf16.h>

typedef __attribute__((ext_vector_type(8)))  _Float16 v8h;
typedef __attribute__((ext_vector_type(16))) _Float16 v16h;
typedef __attribute__((ext_vector_type(8)))  float    v8f;

#define DD 1024
#define EE 8
#define FF 2048

#define XS_STR 1032  // 1024 + 8 f16 pad -> row step 2064B = 516 dwords -> bank step 4 (conflict-free)
#define HS_STR 72    // 64 + 8 f16 pad   -> row step 144B  = 36 dwords  -> bank step 36 (conflict-free)

// ---------------- gating: logits -> top2 -> softmax, plus x fp32 -> f16 ----------------
__global__ __launch_bounds__(256) void gate_kernel(
    const float* __restrict__ x, const float* __restrict__ gw,
    _Float16* __restrict__ xh, int* __restrict__ topi, float* __restrict__ topw)
{
  __shared__ float gws[DD * EE];                 // 32 KB gate weights, block-resident
  for (int i = threadIdx.x; i < DD * EE; i += 256) gws[i] = gw[i];
  __syncthreads();

  const int wave = threadIdx.x >> 5, lane = threadIdx.x & 31;
  const int t = blockIdx.x * 8 + wave;           // one wave32 per token
  const float* xr = x + (size_t)t * DD;
  _Float16* xhr = xh + (size_t)t * DD;

  float acc[EE];
  #pragma unroll
  for (int e = 0; e < EE; ++e) acc[e] = 0.f;

  #pragma unroll 4
  for (int j = 0; j < DD / 32; ++j) {
    int d = j * 32 + lane;
    float xv = xr[d];
    xhr[d] = (_Float16)xv;                       // convert x to f16 while it is in registers
    #pragma unroll
    for (int e = 0; e < EE; ++e) acc[e] += xv * gws[d * EE + e];
  }
  #pragma unroll
  for (int off = 16; off > 0; off >>= 1) {
    #pragma unroll
    for (int e = 0; e < EE; ++e) acc[e] += __shfl_xor(acc[e], off, 32);
  }
  if (lane == 0) {
    int i0 = 0; float v0 = acc[0];
    #pragma unroll
    for (int e = 1; e < EE; ++e) if (acc[e] > v0) { v0 = acc[e]; i0 = e; }
    int i1 = (i0 == 0) ? 1 : 0; float v1 = -3.4e38f;
    #pragma unroll
    for (int e = 0; e < EE; ++e) if (e != i0 && acc[e] > v1) { v1 = acc[e]; i1 = e; }
    float e0 = 1.f;                              // exp(v0 - v0)
    float e1 = __expf(v1 - v0);
    float inv = 1.f / (e0 + e1);
    topi[t * 2 + 0] = i0; topi[t * 2 + 1] = i1;
    topw[t * 2 + 0] = e0 * inv; topw[t * 2 + 1] = e1 * inv;
  }
}

// ---------------- zero output + expert counters ----------------
__global__ void zero_kernel(float* __restrict__ out, size_t n, int* __restrict__ cnts)
{
  size_t i = (size_t)blockIdx.x * blockDim.x + threadIdx.x;
  size_t stride = (size_t)gridDim.x * blockDim.x;
  for (; i < n; i += stride) out[i] = 0.f;
  if (blockIdx.x == 0 && threadIdx.x < EE) cnts[threadIdx.x] = 0;
}

// ---------------- fp32 [e][R][C] -> f16 [e][C][R] tiled transpose-convert ----------------
__global__ __launch_bounds__(256) void transcvt_kernel(
    const float* __restrict__ src, _Float16* __restrict__ dst, int R, int C)
{
  __shared__ float tile[32][33];
  const int e = blockIdx.z;
  src += (size_t)e * R * C;
  dst += (size_t)e * R * C;
  const int c0 = blockIdx.x * 32, r0 = blockIdx.y * 32;
  #pragma unroll
  for (int j = 0; j < 4; ++j) {
    int r = r0 + threadIdx.y + j * 8;
    tile[threadIdx.y + j * 8][threadIdx.x] = src[(size_t)r * C + c0 + threadIdx.x];
  }
  __syncthreads();
  #pragma unroll
  for (int j = 0; j < 4; ++j) {
    int c = c0 + threadIdx.y + j * 8;
    dst[(size_t)c * R + r0 + threadIdx.x] = (_Float16)tile[threadIdx.x][threadIdx.y + j * 8];
  }
}

// ---------------- per-expert token compaction ----------------
__global__ void route_kernel(const int* __restrict__ topi, const float* __restrict__ topw,
                             int* __restrict__ cnts, int* __restrict__ tokL,
                             float* __restrict__ wL, int T)
{
  int t = blockIdx.x * blockDim.x + threadIdx.x;
  if (t >= T) return;
  #pragma unroll
  for (int k = 0; k < 2; ++k) {
    int e = topi[t * 2 + k];
    int slot = atomicAdd(&cnts[e], 1);
    tokL[(size_t)e * T + slot] = t;
    wL[(size_t)e * T + slot] = topw[t * 2 + k];
  }
}

// ---- WMMA fragment loaders (16-bit, wave32 layouts from CDNA5 ISA) ----
// A 16x32: lane<16 holds K{0..7},{16..23}; lane>=16 holds K{8..15},{24..31}
__device__ __forceinline__ v16h ldA16(const _Float16* p) {
  v16h r;
  ((v8h*)&r)[0] = *(const v8h*)(p);
  ((v8h*)&r)[1] = *(const v8h*)(p + 16);
  return r;
}
// B 32x16: lane<16 holds K{0..7},{16..23}; lane>=16 holds K{8..15},{24..31} (caller adds (lane>>4)*8)
__device__ __forceinline__ v16h ldB16(const _Float16* p) {
  v16h r;
  ((v8h*)&r)[0] = *(const v8h*)(p);
  ((v8h*)&r)[1] = *(const v8h*)(p + 16);
  return r;
}

// ---------------- main expert kernel: 64 tokens/WG, 16 waves ----------------
// X tile (64x1024 f16, 129 KB) is LDS-resident for the whole workgroup (CDNA5 WGP
// has 320 KB LDS) -> loaded ONCE; stage 1 then runs barrier-free from LDS.
// stage 1: Hc[64x64] = silu(X @ w1 slice), one 16x16 tile per wave,
//          two independent WMMA accumulation chains (even/odd k-steps)
// stage 2: acc[64x1024] += Hc @ w2 slice; each wave owns a 64-wide D window
__global__ __launch_bounds__(512) void moe_kernel(
    const _Float16* __restrict__ xh,
    const _Float16* __restrict__ w1t,   // [E][F][D]  (K = d contiguous)
    const _Float16* __restrict__ w2t,   // [E][D][F]  (K = f contiguous)
    const int* __restrict__ cnts,
    const int* __restrict__ tokL,
    const float* __restrict__ wL,
    float* __restrict__ out, int T)
{
  __shared__ _Float16 Xs[64 * XS_STR];  // 129 KB resident token tile
  __shared__ _Float16 Hs[64 * HS_STR];  // 9 KB silu(h) chunk (F window of 64)
  __shared__ int   Ts[64];
  __shared__ float Wg[64];

  const int e = blockIdx.x;
  const int cnt = cnts[e];
  const int r0 = blockIdx.y * 64;
  if (r0 >= cnt) return;                 // uniform early exit for over-launched tiles
  const int mcount = min(64, cnt - r0);

  const int tid = threadIdx.x;
  const int wave = tid >> 5, lane = tid & 31;
  const int ln = lane & 15, lh = lane >> 4;

  if (tid < 64) {
    Ts[tid] = (tid < mcount) ? tokL[(size_t)e * T + r0 + tid] : 0;
    Wg[tid] = (tid < mcount) ? wL[(size_t)e * T + r0 + tid] : 0.f;
  }
  __syncthreads();

  // one-time gather of the token tile: Xs[m][0..1023] = xh[Ts[m]][:]
  for (int i = tid; i < 64 * 128; i += 512) {
    int m = i >> 7, c = i & 127;
    *(v8h*)(Xs + m * XS_STR + c * 8) =
        *(const v8h*)(xh + (size_t)Ts[m] * DD + c * 8);
  }
  __syncthreads();

  const _Float16* w1e = w1t + (size_t)e * FF * DD;
  const _Float16* w2e = w2t + (size_t)e * DD * FF;

  const int mb1 = wave >> 2, fb = wave & 3;  // stage-1 tile coords (4x4 waves)
  const int dwin = wave * 64;                // stage-2 D window for this wave

  v8f acc[4][4] = {};                        // 128 VGPRs of f32 accumulators

  for (int fw = 0; fw < FF / 64; ++fw) {
    const int f0 = fw * 64;
    v8f h0 = {}, h1 = {};                    // two independent WMMA chains
    const _Float16* b1p = w1e + (size_t)(f0 + fb * 16 + ln) * DD + lh * 8;
    const _Float16* a1p = Xs + (mb1 * 16 + ln) * XS_STR + lh * 8;

    // warm WGP$ for the NEXT window's weight rows (global_prefetch_b8)
    if (fw + 1 < FF / 64) {
      __builtin_prefetch(b1p + (size_t)64 * DD, 0, 3);
      __builtin_prefetch(w2e + (size_t)(dwin + ln) * FF + f0 + 64, 0, 3);
    }

    // stage 1: barrier-free, K=1024 in 4 blocks of 8 k-steps
    #pragma unroll 1
    for (int dc = 0; dc < 4; ++dc) {
      #pragma unroll
      for (int ks = 0; ks < 8; ks += 2) {
        int k0 = dc * 256 + ks * 32;
        v16h a0 = ldA16(a1p + k0);
        v16h b0 = ldB16(b1p + k0);
        h0 = __builtin_amdgcn_wmma_f32_16x16x32_f16(false, a0, false, b0,
                                                    (short)0, h0, false, false);
        v16h a1 = ldA16(a1p + k0 + 32);
        v16h b1 = ldB16(b1p + k0 + 32);
        h1 = __builtin_amdgcn_wmma_f32_16x16x32_f16(false, a1, false, b1,
                                                    (short)0, h1, false, false);
      }
    }

    // silu in f32 (fast rcp), store f16 chunk to LDS
    #pragma unroll
    for (int v = 0; v < 8; ++v) {
      float hv = h0[v] + h1[v];
      float s = hv * __builtin_amdgcn_rcpf(1.f + __expf(-hv));
      Hs[(mb1 * 16 + lh * 8 + v) * HS_STR + fb * 16 + ln] = (_Float16)s;
    }
    __syncthreads();

    // stage 2: acc += Hc(64xK64) @ W2(K64 x 64-wide window)
    #pragma unroll
    for (int nb = 0; nb < 4; ++nb) {
      const _Float16* b2p = w2e + (size_t)(dwin + nb * 16 + ln) * FF + f0 + lh * 8;
      #pragma unroll
      for (int ks = 0; ks < 2; ++ks) {
        v16h b = ldB16(b2p + ks * 32);
        #pragma unroll
        for (int mb = 0; mb < 4; ++mb) {
          v16h a = ldA16(Hs + (mb * 16 + ln) * HS_STR + ks * 32 + lh * 8);
          acc[mb][nb] = __builtin_amdgcn_wmma_f32_16x16x32_f16(false, a, false, b,
                                                               (short)0, acc[mb][nb],
                                                               false, false);
        }
      }
    }
    __syncthreads();                       // Hs rewritten next iteration
  }

  // epilogue: scale by gate weight, masked native fp32 atomic add (exactly 2 adds
  // per out element across the grid -> bitwise deterministic: 2-operand add commutes)
  #pragma unroll
  for (int mb = 0; mb < 4; ++mb) {
    #pragma unroll
    for (int v = 0; v < 8; ++v) {
      int row = mb * 16 + lh * 8 + v;
      if (row < mcount) {
        float wgt = Wg[row];
        float* orow = out + (size_t)Ts[row] * DD + dwin;
        #pragma unroll
        for (int nb = 0; nb < 4; ++nb)
          unsafeAtomicAdd(orow + nb * 16 + ln, wgt * acc[mb][nb][v]);
      }
    }
  }
}

extern "C" void kernel_launch(void* const* d_in, const int* in_sizes, int n_in,
                              void* d_out, int out_size, void* d_ws, size_t ws_size,
                              hipStream_t stream)
{
  const float* x  = (const float*)d_in[0];
  const float* gw = (const float*)d_in[1];
  const float* w1 = (const float*)d_in[2];
  const float* w2 = (const float*)d_in[3];
  float* out = (float*)d_out;
  const int T = in_sizes[0] / DD;   // 8192 tokens

  // workspace carve-up (~85 MB total)
  char* ws = (char*)d_ws;
  size_t off = 0;
  auto carve = [&](size_t bytes) {
    void* p = ws + off;
    off = (off + bytes + 255) & ~(size_t)255;
    return p;
  };
  _Float16* xh   = (_Float16*)carve((size_t)T * DD * 2);
  _Float16* w1t  = (_Float16*)carve((size_t)EE * FF * DD * 2);
  _Float16* w2t  = (_Float16*)carve((size_t)EE * DD * FF * 2);
  int*      topi = (int*)carve((size_t)T * 2 * 4);
  float*    topw = (float*)carve((size_t)T * 2 * 4);
  int*      cnts = (int*)carve(256);
  int*      tokL = (int*)carve((size_t)EE * T * 4);
  float*    wLs  = (float*)carve((size_t)EE * T * 4);

  zero_kernel<<<2048, 256, 0, stream>>>(out, (size_t)T * DD, cnts);
  gate_kernel<<<T / 8, 256, 0, stream>>>(x, gw, xh, topi, topw);
  transcvt_kernel<<<dim3(FF / 32, DD / 32, EE), dim3(32, 8), 0, stream>>>(w1, w1t, DD, FF);
  transcvt_kernel<<<dim3(DD / 32, FF / 32, EE), dim3(32, 8), 0, stream>>>(w2, w2t, FF, DD);
  route_kernel<<<(T + 255) / 256, 256, 0, stream>>>(topi, topw, cnts, tokL, wLs, T);
  moe_kernel<<<dim3(EE, T / 64), 512, 0, stream>>>(xh, w1t, w2t, cnts, tokL, wLs, out, T);
}
